// cheb_conv_withSAt__3513283248988
// MI455X (gfx1250) — matrix operand order repaired and checked by reading, not compiled, not run on hardware
//
#include <hip/hip_runtime.h>

typedef __attribute__((ext_vector_type(16))) __bf16 v16bf;
typedef __attribute__((ext_vector_type(8)))  __bf16 v8bf;
typedef __attribute__((ext_vector_type(8)))  float  v8f;

#define B_    16
#define N_    1024
#define CIN   64
#define T_    12
#define CT    768     // CIN * T
#define KCH   3
#define COUT  64

// One block per (b, 16-wide n tile). 256 threads = 8 waves (wave32).
__global__ __launch_bounds__(256) void cheb_conv_sat_fused_kernel(
    const float* __restrict__ x,      // [B, N, CIN, T]  (CT contiguous)
    const float* __restrict__ sa,     // [B, N, N]
    const float* __restrict__ cheb,   // [K, N, N]
    const float* __restrict__ theta,  // [K, CIN, COUT]
    float* __restrict__ out)          // [B, N, COUT, T]
{
    // f32 async-copy landing zone (half m-slab), then bf16 operand buffers laid
    // out for contiguous per-lane WMMA reads.
    __shared__ __attribute__((aligned(16))) float  Xf[16][CT];          // 48 KB
    __shared__ __attribute__((aligned(32))) __bf16 Xs[CT][32];          // 48 KB [ct][m]
    __shared__ __attribute__((aligned(32))) __bf16 As[KCH][16][32];     //  3 KB [k][n][m]
    __shared__ __attribute__((aligned(32))) __bf16 Rs[T_][16][CIN];     // 24 KB [t][n][c]
    __shared__ __attribute__((aligned(32))) __bf16 Ts[KCH][COUT][CIN];  // 24 KB [k][o][c]

    const int tid  = threadIdx.x;
    const int wave = tid >> 5;
    const int lane = tid & 31;
    const int lrow = lane & 15;        // row (A) / column (B,D) index within half-wave
    const int hi   = lane >> 4;        // 0: lanes 0-15, 1: lanes 16-31
    const int ko   = hi ? 16 : 0;      // A-operand K sub-offset per half-wave

    const int b  = blockIdx.x >> 6;
    const int n0 = (blockIdx.x & 63) << 4;

    const float* xb  = x  + (size_t)b * N_ * CT;
    const float* sab = sa + (size_t)b * N_ * N_;

    // Stage all theta_k transposed into Ts[k][o][c] once (12288 elements).
    for (int i = tid; i < KCH * CIN * COUT; i += 256) {
        const int k = i >> 12;
        const int c = (i >> 6) & 63;
        const int o = i & 63;
        Ts[k][o][c] = (__bf16)theta[i];
    }

    // Stage-1 accumulators: [k][ct-tile], 18 tiles x 8 VGPRs.
    v8f acc1[KCH][6];
#pragma unroll
    for (int k = 0; k < KCH; ++k)
#pragma unroll
        for (int i = 0; i < 6; ++i)
            acc1[k][i] = (v8f){0.f,0.f,0.f,0.f,0.f,0.f,0.f,0.f};

    for (int ms = 0; ms < N_ / 32; ++ms) {
        const int m = ms * 32;
        __syncthreads();   // previous step's operand reads complete before overwrite

        // A tiles for all k: As[k][n][mm] = bf16(cheb[k][m+mm][n0+n] * sa[b][m+mm][n0+n])
        {
            const int nn = tid & 15;
            const int mm = (tid >> 4) * 2;      // 0,2,...,30
            const size_t r0 = (size_t)(m + mm) * N_ + (n0 + nn);
            const float s0 = sab[r0];
            const float s1 = sab[r0 + N_];
#pragma unroll
            for (int k = 0; k < KCH; ++k) {
                const float* chk = cheb + (size_t)k * N_ * N_;
                union { __bf16 h[2]; unsigned u; } pk;
                pk.h[0] = (__bf16)(chk[r0]      * s0);
                pk.h[1] = (__bf16)(chk[r0 + N_] * s1);
                *(unsigned*)&As[k][nn][mm] = pk.u;   // packed bf16x2 store
            }
        }

        // X slab in two 16-row halves: async copy f32 -> LDS, then bf16 transpose.
#pragma unroll
        for (int half = 0; half < 2; ++half) {
            const int mb = m + half * 16;
            // 16 x 768 f32 = 3072 float4 transfers, 12 per thread (ASYNCcnt-tracked).
#pragma unroll
            for (int it = 0; it < 12; ++it) {
                const int q  = tid + it * 256;       // 0..3071
                const int mm = q / 192;
                const int c4 = (q % 192) * 4;
                const float* gp = xb + (size_t)(mb + mm) * CT + c4;
                const unsigned ldst = (unsigned)(uintptr_t)&Xf[mm][c4];
                asm volatile("global_load_async_to_lds_b128 %0, %1, off"
                             :: "v"(ldst), "v"(gp) : "memory");
            }
            asm volatile("s_wait_asynccnt 0" ::: "memory");
            __syncthreads();

            // Transpose + convert: Xs[ct][half*16 + mm] = bf16(Xf[mm][ct]),
            // packing adjacent-m bf16 pairs into single b32 LDS stores.
#pragma unroll
            for (int it = 0; it < 6; ++it) {
                const int q   = tid + it * 256;      // 0..1535
                const int mm0 = (q / 192) * 2;
                const int c4  = (q % 192) * 4;
                const float4 a = *(const float4*)&Xf[mm0][c4];
                const float4 c = *(const float4*)&Xf[mm0 + 1][c4];
                union { __bf16 h[2]; unsigned u; } pk;
                pk.h[0] = (__bf16)a.x; pk.h[1] = (__bf16)c.x;
                *(unsigned*)&Xs[c4 + 0][half * 16 + mm0] = pk.u;
                pk.h[0] = (__bf16)a.y; pk.h[1] = (__bf16)c.y;
                *(unsigned*)&Xs[c4 + 1][half * 16 + mm0] = pk.u;
                pk.h[0] = (__bf16)a.z; pk.h[1] = (__bf16)c.z;
                *(unsigned*)&Xs[c4 + 2][half * 16 + mm0] = pk.u;
                pk.h[0] = (__bf16)a.w; pk.h[1] = (__bf16)c.w;
                *(unsigned*)&Xs[c4 + 3][half * 16 + mm0] = pk.u;
            }
            __syncthreads();   // Xf free for next half; Xs visible
        }

        if (m + 32 < N_) {
            __builtin_prefetch(sab + (size_t)(m + 32) * N_ + n0 + lrow, 0, 1);
        }

        // A operands for all k (shared across the 8 waves):
        // lanes 0-15: K {0..15}; lanes 16-31: K {16..31}
        union { v16bf v; v8bf h[2]; } au[KCH];
#pragma unroll
        for (int k = 0; k < KCH; ++k) {
            au[k].h[0] = *(const v8bf*)&As[k][lrow][ko];
            au[k].h[1] = *(const v8bf*)&As[k][lrow][8 + ko];
        }

        // Each B tile (x) feeds 3 WMMAs (one per k): 3x operand reuse.
#pragma unroll
        for (int tt = 0; tt < 6; ++tt) {
            const int ct0 = (wave * 6 + tt) << 4;
            const v16bf bv = *(const v16bf*)&Xs[ct0 + lrow][hi ? 16 : 0];
#pragma unroll
            for (int k = 0; k < KCH; ++k) {
                acc1[k][tt] = __builtin_amdgcn_wmma_f32_16x16x32_bf16(
                    false, au[k].v, false, bv, (short)0, acc1[k][tt], false, false);
            }
        }
    }

    // Stage 2: out_t[n, o] = sum_k rhs_k,t[n, :64] @ theta_k[:64, o]
    v8f acc2[6];
#pragma unroll
    for (int i = 0; i < 6; ++i) acc2[i] = (v8f){0.f,0.f,0.f,0.f,0.f,0.f,0.f,0.f};

    for (int k = 0; k < KCH; ++k) {
        __syncthreads();   // previous k's Rs reads complete
        // Scatter rhs_k to Rs[t][n][c]. D layout: VGPR r -> row n = r + 8*hi.
#pragma unroll
        for (int tt = 0; tt < 6; ++tt) {
            const int ct = ((wave * 6 + tt) << 4) + lrow;
            const int c  = ct / T_;
            const int t  = ct % T_;
#pragma unroll
            for (int r = 0; r < 8; ++r) {
                Rs[t][r + hi * 8][c] = (__bf16)acc1[k][tt][r];
            }
        }
        __syncthreads();

        // Each wave owns 6 of the 48 (t, o-tile) outputs; K=64 -> 2 WMMAs.
#pragma unroll
        for (int tt = 0; tt < 6; ++tt) {
            const int g  = wave * 6 + tt;
            const int t  = g >> 2;
            const int o0 = (g & 3) << 4;
#pragma unroll
            for (int s = 0; s < 2; ++s) {
                union { v16bf v; v8bf h[2]; } a2;
                a2.h[0] = *(const v8bf*)&Rs[t][lrow][s * 32 + ko];
                a2.h[1] = *(const v8bf*)&Rs[t][lrow][s * 32 + 8 + ko];
                const v16bf b2 = *(const v16bf*)&Ts[k][o0 + lrow][s * 32 + (hi ? 16 : 0)];
                acc2[tt] = __builtin_amdgcn_wmma_f32_16x16x32_bf16(
                    false, a2.v, false, b2, (short)0, acc2[tt], false, false);
            }
        }
    }

    // Epilogue: ReLU + store f32 out[b][n][o][t]
#pragma unroll
    for (int tt = 0; tt < 6; ++tt) {
        const int g = wave * 6 + tt;
        const int t = g >> 2;
        const int o = ((g & 3) << 4) + lrow;
#pragma unroll
        for (int r = 0; r < 8; ++r) {
            const int n = n0 + r + hi * 8;
            const float v = acc2[tt][r];
            out[(((size_t)b * N_ + n) * COUT + o) * T_ + t] = v > 0.f ? v : 0.f;
        }
    }
}

extern "C" void kernel_launch(void* const* d_in, const int* in_sizes, int n_in,
                              void* d_out, int out_size, void* d_ws, size_t ws_size,
                              hipStream_t stream) {
    (void)in_sizes; (void)n_in; (void)d_ws; (void)ws_size; (void)out_size;
    const float* x     = (const float*)d_in[0];
    const float* sa    = (const float*)d_in[1];
    const float* cheb  = (const float*)d_in[2];
    const float* theta = (const float*)d_in[3];
    float* out = (float*)d_out;
    // 16 batches x 64 n-tiles = 1024 blocks, 256 threads (8 wave32 waves)
    cheb_conv_sat_fused_kernel<<<dim3(B_ * (N_ / 16)), dim3(256), 0, stream>>>(
        x, sa, cheb, theta, out);
}
